// Codebook_10797547782671
// MI455X (gfx1250) — compile-verified, matrix operand order checked by reading, not run on hardware
//
#include <hip/hip_runtime.h>
#include <hip/hip_bf16.h>
#include <stdint.h>

typedef unsigned short u16;
typedef __attribute__((ext_vector_type(16))) __bf16 v16bf;
typedef __attribute__((ext_vector_type(8)))  float  v8f;
typedef __attribute__((ext_vector_type(8)))  unsigned int v8u;
typedef __attribute__((ext_vector_type(4)))  int v4i;

#define N_ROWS   32768   // 32 * 32 * 32 latent vectors
#define DDIM     256     // latent dim (GEMM K)
#define KCB      1024    // codebook entries (GEMM N)
#define ZQ_ELEMS 8388608 // 32*256*32*32

// ---- workspace layout (bytes) ----
#define ZF_OFF   0u                 // bf16 zf [32768][256]  = 16,777,216 B
#define CB_OFF   16777216u          // bf16 cb [1024][256]   =    524,288 B
#define CN_OFF   17301504u          // f32 cnorm[1024]       =      4,096 B
#define PART_OFF 17305600u          // f32 partials[32768]   =    131,072 B

// LDS slab geometry: 16 codewords x (256 + 8 pad) halfs -> 528B col stride
#define COLSTRIDE 264
#define BUF_HALFS (16 * COLSTRIDE)

#if __has_builtin(__builtin_amdgcn_global_load_async_to_lds_b128) && \
    __has_builtin(__builtin_amdgcn_s_wait_asynccnt)
#define HAVE_ASYNC_LDS 1
#else
#define HAVE_ASYNC_LDS 0
#endif

__device__ __forceinline__ u16 f2bf(float x) {
    // round-to-nearest-even fp32 -> bf16 (inputs are finite)
    unsigned u = __float_as_uint(x);
    unsigned r = u + 0x7FFFu + ((u >> 16) & 1u);
    return (u16)(r >> 16);
}

#if HAVE_ASYNC_LDS
__device__ __forceinline__ void async_cp_b128(const u16* gsrc, u16* lds_generic) {
    // Builtin signature (from probe): (V4i addrspace(1)*, V4i addrspace(3)*, Ii, Ii)
    // LDS offset = low 32 bits of the generic pointer (ISA 10.2 truncation rule)
    __attribute__((address_space(1))) v4i* g =
        (__attribute__((address_space(1))) v4i*)(uintptr_t)gsrc;
    __attribute__((address_space(3))) v4i* l =
        (__attribute__((address_space(3))) v4i*)(unsigned)(uintptr_t)lds_generic;
    __builtin_amdgcn_global_load_async_to_lds_b128(g, l, 0, 0);
}
#endif

// ---------------------------------------------------------------------------
// Kernel 1: z NCHW (f32) -> zf row-major [n][c] (bf16). 32x32 LDS tile.
// grid (hwTiles=32, cTiles=8, b=32), block (32, 8)
// ---------------------------------------------------------------------------
__global__ __launch_bounds__(256) void transpose_to_bf16(
    const float* __restrict__ z, u16* __restrict__ zf) {
    __shared__ float tile[32][33];
    const int b   = blockIdx.z;
    const int c0  = blockIdx.y * 32;
    const int hw0 = blockIdx.x * 32;
    const int tx = threadIdx.x, ty = threadIdx.y;
#pragma unroll
    for (int r = 0; r < 4; ++r) {
        const int cl = ty + r * 8;
        tile[cl][tx] = z[((b * 256 + c0 + cl) * 1024) + hw0 + tx];
    }
    __syncthreads();
#pragma unroll
    for (int r = 0; r < 4; ++r) {
        const int hwl = ty + r * 8;
        zf[(size_t)(b * 1024 + hw0 + hwl) * DDIM + c0 + tx] = f2bf(tile[tx][hwl]);
    }
}

// ---------------------------------------------------------------------------
// Kernel 2: codebook f32 -> bf16 copy + cnorm[k] = ||c_k||^2 (fp32).
// One wave per codeword row. grid 128, block 256 (8 waves).
// ---------------------------------------------------------------------------
__global__ __launch_bounds__(256) void cb_prep(
    const float* __restrict__ cb, u16* __restrict__ cbb, float* __restrict__ cnorm) {
    const int lane = threadIdx.x & 31;
    const int wave = threadIdx.x >> 5;
    const int row  = blockIdx.x * 8 + wave;
    const float* src = cb + (size_t)row * DDIM + lane * 8;
    float4 f0 = *(const float4*)src;
    float4 f1 = *(const float4*)(src + 4);
    float ssq = f0.x*f0.x + f0.y*f0.y + f0.z*f0.z + f0.w*f0.w
              + f1.x*f1.x + f1.y*f1.y + f1.z*f1.z + f1.w*f1.w;
    uint4 o;
    o.x = (unsigned)f2bf(f0.x) | ((unsigned)f2bf(f0.y) << 16);
    o.y = (unsigned)f2bf(f0.z) | ((unsigned)f2bf(f0.w) << 16);
    o.z = (unsigned)f2bf(f1.x) | ((unsigned)f2bf(f1.y) << 16);
    o.w = (unsigned)f2bf(f1.z) | ((unsigned)f2bf(f1.w) << 16);
    *(uint4*)(cbb + (size_t)row * DDIM + lane * 8) = o;
#pragma unroll
    for (int m = 16; m >= 1; m >>= 1) ssq += __shfl_xor(ssq, m, 32);
    if (lane == 0) cnorm[row] = ssq;
}

// ---------------------------------------------------------------------------
// Kernel 3: fused bf16 WMMA GEMM + row-wise argmin of (||c||^2 - 2 z.c).
// Block = 256 threads = 8 waves, each wave owns 16 rows (block: 128 rows).
// 64 column tiles of 16 codewords; codebook slab double-buffered in LDS via
// async-to-LDS copies (ASYNCcnt), one barrier per tile, copy overlapped with
// the 8 WMMAs of the previous tile. grid 256, block 256.
// ---------------------------------------------------------------------------
__global__ __launch_bounds__(256) void vq_argmin_kernel(
    const u16* __restrict__ zf, const u16* __restrict__ cbb,
    const float* __restrict__ cnorm, float* __restrict__ outIdx) {
#if HAVE_ASYNC_LDS
    __shared__ __align__(16) u16 smem[2 * BUF_HALFS];
#else
    __shared__ __align__(16) u16 smem[BUF_HALFS];
#endif

    const int lane    = threadIdx.x & 31;
    const int wave    = threadIdx.x >> 5;
    const int l15     = lane & 15;
    const int halfSel = lane >> 4;           // 0: lanes 0-15, 1: lanes 16-31
    const int rowBase = blockIdx.x * 128 + wave * 16;

    // ---- load A fragments for this wave's 16 rows, full K=256, once ----
    v16bf a[8];
    {
        const u16* arow = zf + (size_t)(rowBase + l15) * DDIM;
#pragma unroll
        for (int f = 0; f < 8; ++f) {
            const int k0 = f * 32 + halfSel * 8;
            v8u r;
            *(uint4*)&r       = *(const uint4*)(arow + k0);       // K k0..k0+7
            *((uint4*)&r + 1) = *(const uint4*)(arow + k0 + 16);  // K k0+16..k0+23
            a[f] = __builtin_bit_cast(v16bf, r);
        }
    }

    float minv[8];
    int   mini[8];
#pragma unroll
    for (int r = 0; r < 8; ++r) { minv[r] = 3.4e38f; mini[r] = 0; }

    const int copyCol = threadIdx.x >> 4;    // 0..15 (codeword within tile)
    const int copySeg = threadIdx.x & 15;    // 0..15 (32B segment of K row)
    const u16* copySrc0 = cbb + (size_t)copyCol * DDIM + copySeg * 16;
    u16* copyDst0 = smem + copyCol * COLSTRIDE + copySeg * 16;

#if HAVE_ASYNC_LDS
    // prefill buffer 0 with tile 0
    async_cp_b128(copySrc0,     copyDst0);
    async_cp_b128(copySrc0 + 8, copyDst0 + 8);
#endif

    for (int t = 0; t < 64; ++t) {
#if HAVE_ASYNC_LDS
        __builtin_amdgcn_s_wait_asynccnt(0);   // my copies for tile t landed
        __syncthreads();                       // all waves' copies + compute(t-1)
        if (t < 63) {                          // overlap copy(t+1) with compute(t)
            const u16* src = copySrc0 + (size_t)(t + 1) * 16 * DDIM;
            u16* dst = copyDst0 + ((t + 1) & 1) * BUF_HALFS;
            async_cp_b128(src,     dst);
            async_cp_b128(src + 8, dst + 8);
        }
        const u16* bufBase = smem + (t & 1) * BUF_HALFS;
#else
        __syncthreads();                       // previous tile's compute done
        {   // stage 16 codewords x 256 K (bf16) into LDS (8 KB, 32B/thread)
            const u16* src = copySrc0 + (size_t)t * 16 * DDIM;
            uint4 d0 = *(const uint4*)src;
            uint4 d1 = *(const uint4*)(src + 8);
            *(uint4*)copyDst0       = d0;
            *((uint4*)copyDst0 + 1) = d1;
        }
        __syncthreads();
        const u16* bufBase = smem;
#endif

        const int   col = t * 16 + l15;
        const float cn  = cnorm[col];

        v8f acc = {0.f, 0.f, 0.f, 0.f, 0.f, 0.f, 0.f, 0.f};
        const u16* bcol = bufBase + l15 * COLSTRIDE + halfSel * 8;
#pragma unroll
        for (int f = 0; f < 8; ++f) {
            v8u r;
            *(uint4*)&r       = *(const uint4*)(bcol + f * 32);
            *((uint4*)&r + 1) = *(const uint4*)(bcol + f * 32 + 16);
            v16bf b = __builtin_bit_cast(v16bf, r);
            acc = __builtin_amdgcn_wmma_f32_16x16x32_bf16(
                false, a[f], false, b, (short)0, acc, false, false);
        }
#pragma unroll
        for (int r = 0; r < 8; ++r) {
            const float s = fmaf(-2.f, acc[r], cn);   // ||c||^2 - 2 z.c
            if (s < minv[r]) { minv[r] = s; mini[r] = col; }
        }
    }

    // ---- cross-lane argmin within each 16-lane half (C-layout: cols on lanes)
#pragma unroll
    for (int r = 0; r < 8; ++r) {
        float v = minv[r];
        int   i = mini[r];
#pragma unroll
        for (int m = 8; m >= 1; m >>= 1) {
            const float v2 = __shfl_xor(v, m, 32);
            const int   i2 = __shfl_xor(i, m, 32);
            if (v2 < v || (v2 == v && i2 < i)) { v = v2; i = i2; }
        }
        minv[r] = v; mini[r] = i;
    }
    if (l15 == 0) {
        // lanes 0-15 hold rows rowBase+0..7, lanes 16-31 rows rowBase+8..15
#pragma unroll
        for (int r = 0; r < 8; ++r)
            outIdx[rowBase + halfSel * 8 + r] = (float)mini[r];
    }
}

// ---------------------------------------------------------------------------
// Kernel 4: gather codebook rows -> z_q (NCHW) + per-block loss partials.
// grid 32768, block 256; one thread per output element.
// ---------------------------------------------------------------------------
__global__ __launch_bounds__(256) void quantize_out(
    const float* __restrict__ z, const float* __restrict__ cbf,
    const float* __restrict__ idxF, float* __restrict__ zq,
    float* __restrict__ partials) {
    const int tid = blockIdx.x * 256 + threadIdx.x;
    const int hw = tid & 1023;
    const int bc = tid >> 10;
    const int c  = bc & 255;
    const int b  = bc >> 8;
    const int n  = b * 1024 + hw;
    const int k  = (int)idxF[n];
    const float v  = cbf[(size_t)k * DDIM + c];
    const float zv = z[tid];
    zq[tid] = v;                               // STE forward == quantized value
    const float d = v - zv;
    float p = d * d;
#pragma unroll
    for (int m = 16; m >= 1; m >>= 1) p += __shfl_xor(p, m, 32);
    __shared__ float wsum[8];
    const int lane = threadIdx.x & 31, wave = threadIdx.x >> 5;
    if (lane == 0) wsum[wave] = p;
    __syncthreads();
    if (threadIdx.x == 0) {
        float s = 0.f;
#pragma unroll
        for (int i = 0; i < 8; ++i) s += wsum[i];
        partials[blockIdx.x] = s;
    }
}

// ---------------------------------------------------------------------------
// Kernel 5: deterministic final loss reduction (fixed order, no atomics).
// loss = (1 + BETA) * mean(diff^2) = 1.25 * sum / 8388608
// ---------------------------------------------------------------------------
__global__ __launch_bounds__(256) void loss_final(
    const float* __restrict__ partials, float* __restrict__ lossOut) {
    __shared__ float ls[256];
    float s = 0.f;
    for (int i = 0; i < 128; ++i) s += partials[threadIdx.x + 256 * i];
    ls[threadIdx.x] = s;
    __syncthreads();
    if (threadIdx.x == 0) {
        float t = 0.f;
        for (int i = 0; i < 256; ++i) t += ls[i];
        *lossOut = 1.25f * t / 8388608.0f;
    }
}

extern "C" void kernel_launch(void* const* d_in, const int* in_sizes, int n_in,
                              void* d_out, int out_size, void* d_ws, size_t ws_size,
                              hipStream_t stream) {
    const float* z  = (const float*)d_in[0];   // [32,256,32,32] f32
    const float* cb = (const float*)d_in[1];   // [1024,256]     f32

    float* out   = (float*)d_out;
    float* zq    = out;                        // 8,388,608 floats (NCHW)
    float* idxF  = out + ZQ_ELEMS;             // 32,768 floats
    float* lossp = out + ZQ_ELEMS + N_ROWS;    // 1 float

    char* ws = (char*)d_ws;
    u16*   zf_bf    = (u16*)(ws + ZF_OFF);
    u16*   cb_bf    = (u16*)(ws + CB_OFF);
    float* cnorm    = (float*)(ws + CN_OFF);
    float* partials = (float*)(ws + PART_OFF);

    transpose_to_bf16<<<dim3(32, 8, 32), dim3(32, 8), 0, stream>>>(z, zf_bf);
    cb_prep<<<128, 256, 0, stream>>>(cb, cb_bf, cnorm);
    vq_argmin_kernel<<<256, 256, 0, stream>>>(zf_bf, cb_bf, cnorm, idxF);
    quantize_out<<<32768, 256, 0, stream>>>(z, cb, idxF, zq, partials);
    loss_final<<<1, 256, 0, stream>>>(partials, lossp);
}
